// Multihead_83330955477774
// MI455X (gfx1250) — matrix operand order repaired and checked by reading, not compiled, not run on hardware
//
#include <hip/hip_runtime.h>

typedef __attribute__((ext_vector_type(16))) _Float16 v16h;
typedef __attribute__((ext_vector_type(8)))  _Float16 h8;
typedef __attribute__((ext_vector_type(8)))  float    v8f;

#define HEADS 16
#define SEQ   2048
#define DIMS  1024
#define HD    64
#define BATCH 2

__device__ __forceinline__ v8f wmma_f16(const v16h& a, const v16h& b, const v8f& c) {
  return __builtin_amdgcn_wmma_f32_16x16x32_f16(false, a, false, b, (short)0, c, false, false);
}

// A fragment (16x32 f16): lane = row; halves 0-15 = K {0-15 | 16-31}
// contiguous.  base = row_ptr + k0 + (hi?16:0), 16B aligned.
__device__ __forceinline__ v16h load_afrag(const _Float16* base) {
  const h8* p = (const h8*)base;
  const h8 lo = p[0];
  const h8 up = p[1];   // +8 halves
  v16h r;
  #pragma unroll
  for (int t = 0; t < 8; ++t) { r[t] = lo[t]; r[8 + t] = up[t]; }
  return r;
}

// Async-copy one 16-byte packet global -> LDS (per-lane), ASYNCcnt tracked.
__device__ __forceinline__ void async_b128(const _Float16* g, const _Float16* lds) {
  const unsigned            lo = (unsigned)(uintptr_t)lds;           // LDS byte offset
  const unsigned long long  ga = (unsigned long long)(uintptr_t)g;   // global address
  asm volatile("global_load_async_to_lds_b128 %0, %1, off"
               :: "v"(lo), "v"(ga) : "memory");
}
__device__ __forceinline__ void async_wait0() {
  asm volatile("s_wait_asynccnt 0" ::: "memory");
}

// Stage the block-shared 16xDIMS f16 A panel into LDS (32 KB) with async loads.
// All 8 waves of the block share the same m0, so this is fetched once.
__device__ __forceinline__ void stage_a_panel(const _Float16* __restrict__ src,
                                              _Float16 (*Ax)[DIMS],
                                              int m0b, int tid) {
  #pragma unroll
  for (int it = 0; it < 8; ++it) {
    const int idx  = it * 256 + tid;   // 2048 x 16B packets
    const int row  = idx >> 7;
    const int col8 = idx & 127;
    async_b128(src + (size_t)(m0b + row) * DIMS + col8 * 8, &Ax[row][col8 * 8]);
  }
  async_wait0();
  __syncthreads();
}

// ---------------------------------------------------------------------------
// Kernel 0: f32 -> f16 conversion of x and the four weight matrices.
// ---------------------------------------------------------------------------
__global__ void cvt_f16_kernel(const float* __restrict__ x,
                               const float* __restrict__ Wq,
                               const float* __restrict__ Wk,
                               const float* __restrict__ Wv,
                               const float* __restrict__ Wo,
                               _Float16* __restrict__ xh,
                               _Float16* __restrict__ Wqh,
                               _Float16* __restrict__ Wkh,
                               _Float16* __restrict__ Wvh,
                               _Float16* __restrict__ Woh) {
  const int z = blockIdx.z;
  const float* src = (z == 0) ? x : (z == 1) ? Wq : (z == 2) ? Wk : (z == 3) ? Wv : Wo;
  _Float16*    dst = (z == 0) ? xh : (z == 1) ? Wqh : (z == 2) ? Wkh : (z == 3) ? Wvh : Woh;
  const int n8 = (z == 0) ? (BATCH * SEQ * DIMS / 8) : (DIMS * DIMS / 8);
  const int i = blockIdx.x * blockDim.x + threadIdx.x;
  if (i < n8) {
    const float4* s = (const float4*)src;
    const float4 a = s[2 * i], b = s[2 * i + 1];
    h8 o;
    o[0] = (_Float16)a.x; o[1] = (_Float16)a.y; o[2] = (_Float16)a.z; o[3] = (_Float16)a.w;
    o[4] = (_Float16)b.x; o[5] = (_Float16)b.y; o[6] = (_Float16)b.z; o[7] = (_Float16)b.w;
    *(h8*)(dst + (size_t)i * 8) = o;
  }
}

// ---------------------------------------------------------------------------
// Kernel 1: fused QKV projection (f16 WMMA, f32 acc) + bias + rotary(q,k) /
// transpose(v). blockIdx.z: 0=Q, 1=K, 2=V.  One wave = 16x64 output strip;
// A panel staged once per block via async-to-LDS; B double-buffered.
// ---------------------------------------------------------------------------
__global__ void qkv_rope_kernel(const _Float16* __restrict__ xh,
                                const _Float16* __restrict__ Wqh,
                                const float* __restrict__ bq,
                                const _Float16* __restrict__ Wkh,
                                const _Float16* __restrict__ Wvh,
                                const float* __restrict__ bv,
                                const float* __restrict__ inv_freq,
                                _Float16* __restrict__ qh, _Float16* __restrict__ kh,
                                _Float16* __restrict__ vTh) {
  __shared__ alignas(16) _Float16 Ax[16][DIMS];   // 32 KB block-shared A panel
  const int tid    = threadIdx.x;
  const int lane   = tid & 31;
  const int wave   = tid >> 5;
  const int lane16 = lane & 15;
  const bool hi    = lane >= 16;
  const int mode   = blockIdx.z;
  const _Float16* W    = (mode == 0) ? Wqh : (mode == 1) ? Wkh : Wvh;
  const float*    bias = (mode == 0) ? bq : (mode == 2) ? bv : nullptr;

  const int task = blockIdx.x * 8 + wave;     // 256 m-tiles x 16 n-strips
  const int m0 = (task >> 4) << 4;
  const int n0 = (task & 15) << 6;            // 64-wide strip == one head
  const int m0b = ((blockIdx.x * 8) >> 4) << 4;  // block-uniform (== m0)

  stage_a_panel(xh, Ax, m0b, tid);

  const _Float16* wrow[4];
  #pragma unroll
  for (int nt = 0; nt < 4; ++nt)
    wrow[nt] = W + (size_t)(n0 + nt * 16 + lane16) * DIMS + (hi ? 16 : 0);

  v8f acc[4];
  #pragma unroll
  for (int nt = 0; nt < 4; ++nt)
    #pragma unroll
    for (int j = 0; j < 8; ++j) acc[nt][j] = 0.f;

  v16h bcur[4];
  #pragma unroll
  for (int nt = 0; nt < 4; ++nt) bcur[nt] = *(const v16h*)(wrow[nt]);

  for (int k0 = 0; k0 < DIMS; k0 += 32) {
    const int k1 = (k0 + 32 < DIMS) ? (k0 + 32) : k0;
    v16h bnext[4];
    #pragma unroll
    for (int nt = 0; nt < 4; ++nt) bnext[nt] = *(const v16h*)(wrow[nt] + k1);
    const v16h a = load_afrag(&Ax[lane16][k0 + (hi ? 16 : 0)]);
    #pragma unroll
    for (int nt = 0; nt < 4; ++nt) acc[nt] = wmma_f16(a, bcur[nt], acc[nt]);
    #pragma unroll
    for (int nt = 0; nt < 4; ++nt) bcur[nt] = bnext[nt];
  }

  const int h = n0 >> 6;
  #pragma unroll
  for (int nt = 0; nt < 4; ++nt) {
    const int e = n0 + nt * 16 + lane16;
    const int d = nt * 16 + lane16;
    const float bb = bias ? bias[e] : 0.f;
    #pragma unroll
    for (int j = 0; j < 8; ++j) {
      const int mr = m0 + j + (hi ? 8 : 0);
      const int b  = mr >> 11;
      const int s  = mr & (SEQ - 1);
      float v = acc[nt][j] + bb;
      if (mode < 2) {
        const float partner = __shfl_xor(v, 1, 32);
        const float fr = (float)s * inv_freq[d >> 1];
        float sn, cs;
        __sincosf(fr, &sn, &cs);
        v = (d & 1) ? fmaf(partner, sn, v * cs)
                    : fmaf(v, cs, -partner * sn);
        _Float16* dst = (mode == 0) ? qh : kh;
        dst[((size_t)(b * HEADS + h) * SEQ + s) * HD + d] = (_Float16)v;
      } else {
        vTh[((size_t)(b * HEADS + h) * HD + d) * SEQ + s] = (_Float16)v;
      }
    }
  }
}

// ---------------------------------------------------------------------------
// Kernel 2: flash attention. One wave per 16-row q-tile; 32-key steps.
// Writes masked logits (qk) to d_out as computed; online softmax; PV via WMMA.
// ---------------------------------------------------------------------------
__global__ void attn_kernel(const _Float16* __restrict__ qh,
                            const _Float16* __restrict__ kh,
                            const _Float16* __restrict__ vTh,
                            float* __restrict__ qk,
                            _Float16* __restrict__ attn) {
  __shared__ alignas(16) _Float16 Plds[8][16][32];  // per-wave P staging
  const int lane   = threadIdx.x & 31;
  const int wave   = threadIdx.x >> 5;
  const int lane16 = lane & 15;
  const bool hi    = lane >= 16;

  const int task = blockIdx.x * 8 + wave;          // 4096 tasks
  const int b  = task >> 11;
  const int h  = (task >> 7) & 15;
  const int q0 = (task & 127) << 4;
  const size_t bh = (size_t)(b * HEADS + h);

  const _Float16* qrow = qh + (bh * SEQ + q0 + lane16) * HD;
  v16h qa[2];
  #pragma unroll
  for (int dc = 0; dc < 2; ++dc)
    qa[dc] = load_afrag(qrow + dc * 32 + (hi ? 16 : 0));

  v8f o[4];
  float Mrow[8], Lrow[8];
  #pragma unroll
  for (int dc = 0; dc < 4; ++dc)
    #pragma unroll
    for (int j = 0; j < 8; ++j) o[dc][j] = 0.f;
  #pragma unroll
  for (int j = 0; j < 8; ++j) { Mrow[j] = -__builtin_inff(); Lrow[j] = 0.f; }

  float* qkbh = qk + bh * SEQ * SEQ;
  const _Float16* kbase = kh  + bh * SEQ * HD;
  const _Float16* vbase = vTh + bh * HD * SEQ;

  for (int kp = 0; kp < SEQ; kp += 32) {
    if (kp <= q0 + 15) {
      // prefetch next k/v tiles (speculative; stays inside workspace)
      __builtin_prefetch(kbase + (size_t)(kp + 32 + lane16) * HD, 0, 1);
      __builtin_prefetch(vbase + (size_t)lane16 * SEQ + kp + 32, 0, 1);

      // ---- S = (q*s)(k*s)^T over two 16-key tiles ----
      v8f s0, s1;
      #pragma unroll
      for (int j = 0; j < 8; ++j) { s0[j] = 0.f; s1[j] = 0.f; }
      #pragma unroll
      for (int dc = 0; dc < 2; ++dc) {
        const _Float16* k0p = kbase + (size_t)(kp + lane16) * HD + dc * 32 + (hi ? 16 : 0);
        const v16h kb0 = *(const v16h*)k0p;
        const v16h kb1 = *(const v16h*)(k0p + 16 * HD);
        s0 = wmma_f16(qa[dc], kb0, s0);
        s1 = wmma_f16(qa[dc], kb1, s1);
      }
      // ---- scale + causal mask + store logits + online softmax ----
      float alpha8[8];
      #pragma unroll
      for (int j = 0; j < 8; ++j) {
        const int mr  = q0 + j + (hi ? 8 : 0);
        const int n0g = kp + lane16;
        float v0 = s0[j] * 0.125f + ((n0g      <= mr) ? 0.f : -1e9f);
        float v1 = s1[j] * 0.125f + ((n0g + 16 <= mr) ? 0.f : -1e9f);
        float* row = qkbh + (size_t)mr * SEQ;
        row[n0g]      = v0;
        row[n0g + 16] = v1;
        float tm = fmaxf(v0, v1);
        #pragma unroll
        for (int off = 1; off < 16; off <<= 1) tm = fmaxf(tm, __shfl_xor(tm, off, 32));
        const float newM  = fmaxf(Mrow[j], tm);
        const float alpha = __expf(Mrow[j] - newM);
        const float p0 = __expf(v0 - newM);
        const float p1 = __expf(v1 - newM);
        float rs = p0 + p1;
        #pragma unroll
        for (int off = 1; off < 16; off <<= 1) rs += __shfl_xor(rs, off, 32);
        Lrow[j] = Lrow[j] * alpha + rs;
        Mrow[j] = newM;
        alpha8[j] = alpha;
        Plds[wave][j + (hi ? 8 : 0)][lane16]      = (_Float16)p0;
        Plds[wave][j + (hi ? 8 : 0)][lane16 + 16] = (_Float16)p1;
      }
      #pragma unroll
      for (int dc = 0; dc < 4; ++dc)
        #pragma unroll
        for (int j = 0; j < 8; ++j) o[dc][j] *= alpha8[j];
      // ---- P A-fragment from LDS (DS in-order within wave) ----
      const v16h pa = load_afrag(&Plds[wave][lane16][hi ? 16 : 0]);
      // ---- O += P (16x32) x V (32x64) ----
      #pragma unroll
      for (int dc = 0; dc < 4; ++dc) {
        const _Float16* vp = vbase + (size_t)(dc * 16 + lane16) * SEQ + kp + (hi ? 16 : 0);
        const v16h vb = *(const v16h*)vp;
        o[dc] = wmma_f16(pa, vb, o[dc]);
      }
    } else {
      // fully-masked tile: write -1e9 logits
      #pragma unroll
      for (int j = 0; j < 8; ++j) {
        const int mr = q0 + j + (hi ? 8 : 0);
        float* row = qkbh + (size_t)mr * SEQ;
        row[kp + lane16]      = -1e9f;
        row[kp + 16 + lane16] = -1e9f;
      }
    }
  }

  // ---- normalize and emit attn output [B,S,D] (f16) ----
  #pragma unroll
  for (int j = 0; j < 8; ++j) {
    const int mr = q0 + j + (hi ? 8 : 0);
    const float inv = 1.0f / Lrow[j];
    _Float16* arow = attn + (size_t)(b * SEQ + mr) * DIMS + h * HD;
    #pragma unroll
    for (int dc = 0; dc < 4; ++dc)
      arow[dc * 16 + lane16] = (_Float16)(o[dc][j] * inv);
  }
}

// ---------------------------------------------------------------------------
// Kernel 3: output projection  out[m,e] = sum_d attn[m,d]*Wo[e,d] + bo[e]
// Same async-staged A panel + B double-buffer structure.
// ---------------------------------------------------------------------------
__global__ void oproj_kernel(const _Float16* __restrict__ A,
                             const _Float16* __restrict__ Woh,
                             const float* __restrict__ bo,
                             float* __restrict__ out) {
  __shared__ alignas(16) _Float16 Ax[16][DIMS];
  const int tid    = threadIdx.x;
  const int lane   = tid & 31;
  const int wave   = tid >> 5;
  const int lane16 = lane & 15;
  const bool hi    = lane >= 16;

  const int task = blockIdx.x * 8 + wave;
  const int m0 = (task >> 4) << 4;
  const int n0 = (task & 15) << 6;
  const int m0b = ((blockIdx.x * 8) >> 4) << 4;

  stage_a_panel(A, Ax, m0b, tid);

  const _Float16* wrow[4];
  #pragma unroll
  for (int nt = 0; nt < 4; ++nt)
    wrow[nt] = Woh + (size_t)(n0 + nt * 16 + lane16) * DIMS + (hi ? 16 : 0);

  v8f acc[4];
  #pragma unroll
  for (int nt = 0; nt < 4; ++nt)
    #pragma unroll
    for (int j = 0; j < 8; ++j) acc[nt][j] = 0.f;

  v16h bcur[4];
  #pragma unroll
  for (int nt = 0; nt < 4; ++nt) bcur[nt] = *(const v16h*)(wrow[nt]);

  for (int k0 = 0; k0 < DIMS; k0 += 32) {
    const int k1 = (k0 + 32 < DIMS) ? (k0 + 32) : k0;
    v16h bnext[4];
    #pragma unroll
    for (int nt = 0; nt < 4; ++nt) bnext[nt] = *(const v16h*)(wrow[nt] + k1);
    const v16h a = load_afrag(&Ax[lane16][k0 + (hi ? 16 : 0)]);
    #pragma unroll
    for (int nt = 0; nt < 4; ++nt) acc[nt] = wmma_f16(a, bcur[nt], acc[nt]);
    #pragma unroll
    for (int nt = 0; nt < 4; ++nt) bcur[nt] = bnext[nt];
  }

  #pragma unroll
  for (int nt = 0; nt < 4; ++nt) {
    const float bb = bo[n0 + nt * 16 + lane16];
    #pragma unroll
    for (int j = 0; j < 8; ++j)
      out[(size_t)(m0 + j + (hi ? 8 : 0)) * DIMS + n0 + nt * 16 + lane16] = acc[nt][j] + bb;
  }
}

// ---------------------------------------------------------------------------
extern "C" void kernel_launch(void* const* d_in, const int* in_sizes, int n_in,
                              void* d_out, int out_size, void* d_ws, size_t ws_size,
                              hipStream_t stream) {
  const float* x        = (const float*)d_in[0];
  // d_in[1] = mask: causal mask computed analytically in-kernel
  const float* Wq       = (const float*)d_in[2];
  const float* bq       = (const float*)d_in[3];
  const float* Wk       = (const float*)d_in[4];
  const float* Wv       = (const float*)d_in[5];
  const float* bv       = (const float*)d_in[6];
  const float* Wo       = (const float*)d_in[7];
  const float* bo       = (const float*)d_in[8];
  const float* inv_freq = (const float*)d_in[9];

  float* out = (float*)d_out;                       // [B,S,D]
  float* qk  = out + (size_t)BATCH * SEQ * DIMS;    // [B,H,S,S]

  const size_t xelems  = (size_t)BATCH * SEQ * DIMS;        // 4M
  const size_t welems  = (size_t)DIMS * DIMS;               // 1M
  const size_t hqelems = (size_t)BATCH * HEADS * SEQ * HD;  // 4M

  _Float16* xh   = (_Float16*)d_ws;
  _Float16* Wqh  = xh   + xelems;
  _Float16* Wkh  = Wqh  + welems;
  _Float16* Wvh  = Wkh  + welems;
  _Float16* Woh  = Wvh  + welems;
  _Float16* qh   = Woh  + welems;
  _Float16* kh   = qh   + hqelems;
  _Float16* vTh  = kh   + hqelems;
  _Float16* attn = vTh  + hqelems;                  // total 48 MB workspace

  cvt_f16_kernel<<<dim3(2048, 1, 5), 256, 0, stream>>>(x, Wq, Wk, Wv, Wo,
                                                       xh, Wqh, Wkh, Wvh, Woh);
  qkv_rope_kernel<<<dim3(512, 1, 3), 256, 0, stream>>>(xh, Wqh, bq, Wkh, Wvh, bv,
                                                       inv_freq, qh, kh, vTh);
  attn_kernel<<<dim3(512), 256, 0, stream>>>(qh, kh, vTh, qk, attn);
  oproj_kernel<<<dim3(512), 256, 0, stream>>>(attn, Woh, bo, out);
}
